// LSTMSeq2seq_81080392614265
// MI455X (gfx1250) — matrix operand order repaired and checked by reading, not compiled, not run on hardware
//
#include <hip/hip_runtime.h>
#include <hip/hip_bf16.h>
#include <stdint.h>

// ---------------- problem constants ----------------
#define BB   32
#define SS   64
#define TT   64
#define EE   512
#define HH   512
#define HDD  1024      // decoder hidden = 2H
#define VV   32000
#define G4H  2048      // 4*H
#define G4HD 4096      // 4*HD
#define DST  1024      // 2*H  (src_states feature dim)
#define ACD  2048      // 2*HD (concat(h,ctx) dim)
#define MROWS 2016     // B*(T-1)

typedef __attribute__((ext_vector_type(16))) __bf16 bf16x16;
typedef __attribute__((ext_vector_type(8)))  float  floatx8;
typedef __attribute__((ext_vector_type(4))) unsigned int uint4v;
typedef __attribute__((ext_vector_type(8))) int          int8v;
typedef __attribute__((ext_vector_type(4))) int          int4v;
typedef unsigned short u16;

__device__ __forceinline__ u16 f2bf(float f) {
  unsigned int u = __float_as_uint(f);
  u += 0x7FFFu + ((u >> 16) & 1u);          // round-to-nearest-even
  return (u16)(u >> 16);
}
__device__ __forceinline__ float sigf(float x) { return 1.0f / (1.0f + __expf(-x)); }

// A fragment (16x32 bf16): lane holds row m = m0+(lane&15); 16 contiguous
// K values starting at 16*(lane>>4).
__device__ __forceinline__ bf16x16 load_fragA(const u16* p) {
  bf16x16 v;
  ((uint4*)&v)[0] = *(const uint4*)(p);
  ((uint4*)&v)[1] = *(const uint4*)(p + 8);
  return v;
}
// B fragment (32x16 bf16) from row-major W(N,K): lane holds column
// n = n0+(lane&15); 16 contiguous K values starting at 16*(lane>>4).
__device__ __forceinline__ bf16x16 load_fragB(const u16* p) {
  bf16x16 v;
  ((uint4*)&v)[0] = *(const uint4*)(p);
  ((uint4*)&v)[1] = *(const uint4*)(p + 8);
  return v;
}

// ---------------- generic bf16 GEMM: C = A(M,K) * W(N,K)^T (+add)(+bias) ----
// block = 256 thr = 8 waves; wave (w) -> subtile m0=(w&1)*16, n0=(w>>1)*64.
// grid = (N/256, M/32).
__global__ __launch_bounds__(256) void gemm_bf16_kernel(
    const u16* __restrict__ A, int lda,
    const u16* __restrict__ W, int ldw,
    float* __restrict__ C, int ldc,
    const float* __restrict__ add, int add_stride,
    const float* __restrict__ bias1, const float* __restrict__ bias2,
    int K)
{
  const int lane = threadIdx.x & 31;
  const int wv   = threadIdx.x >> 5;
  const int lr   = lane & 15;
  const int hi   = lane >> 4;
  const int m0   = blockIdx.y * 32 + (wv & 1) * 16;
  const int n0   = blockIdx.x * 256 + (wv >> 1) * 64;

  const u16* Ap  = A + (size_t)(m0 + lr) * lda + hi * 16;
  const u16* Wp0 = W + (size_t)(n0 +  0 + lr) * ldw + hi * 16;
  const u16* Wp1 = W + (size_t)(n0 + 16 + lr) * ldw + hi * 16;
  const u16* Wp2 = W + (size_t)(n0 + 32 + lr) * ldw + hi * 16;
  const u16* Wp3 = W + (size_t)(n0 + 48 + lr) * ldw + hi * 16;

  floatx8 acc[4] = {};

  for (int k0 = 0; k0 < K; k0 += 32) {
    bf16x16 a  = load_fragA(Ap  + k0);
    bf16x16 b0 = load_fragB(Wp0 + k0);
    bf16x16 b1 = load_fragB(Wp1 + k0);
    bf16x16 b2 = load_fragB(Wp2 + k0);
    bf16x16 b3 = load_fragB(Wp3 + k0);
    __builtin_prefetch(Wp0 + k0 + 256, 0, 0);
    __builtin_prefetch(Ap  + k0 + 256, 0, 0);
    acc[0] = __builtin_amdgcn_wmma_f32_16x16x32_bf16(false, a, false, b0, (short)0, acc[0], false, false);
    acc[1] = __builtin_amdgcn_wmma_f32_16x16x32_bf16(false, a, false, b1, (short)0, acc[1], false, false);
    acc[2] = __builtin_amdgcn_wmma_f32_16x16x32_bf16(false, a, false, b2, (short)0, acc[2], false, false);
    acc[3] = __builtin_amdgcn_wmma_f32_16x16x32_bf16(false, a, false, b3, (short)0, acc[3], false, false);
  }

#pragma unroll
  for (int i = 0; i < 4; ++i) {
    const int n = n0 + 16 * i + lr;
    float bs = 0.0f;
    if (bias1) bs += bias1[n];
    if (bias2) bs += bias2[n];
#pragma unroll
    for (int r = 0; r < 8; ++r) {
      const int m = m0 + hi * 8 + r;
      float v = acc[i][r] + bs;
      if (add) v += add[(size_t)m * add_stride + n];
      C[(size_t)m * ldc + n] = v;
    }
  }
}

// -------- Tensor Data Mover: DMA a 256(N) x 32(K) bf16 W tile into LDS ------
// D# per CDNA5 ISA: group0 {count, lds_addr, global_addr,
// type=2}; group1 {data_size=2B, pad 4 dwords every 16 dwords -> 80B row pitch,
// tensor_dim0=2048, tensor_dim1=32000, tile_dim0=32, tile_dim1=256,
// tensor_dim0_stride=2048}.  2D tensor -> remaining groups zero.
#define WTILE_PITCH 40   // u16 per LDS row (64B data + 16B pad)
__device__ __forceinline__ void tdm_load_w_tile(const u16* gsrc, unsigned lds_off) {
  uint64_t ga = (uint64_t)(uintptr_t)gsrc;
  uint4v g0;
  g0.x = 1u;                                                 // count=1 (valid)
  g0.y = lds_off;                                            // lds_addr (bytes)
  g0.z = (unsigned)(ga & 0xFFFFFFFFu);                       // global_addr[31:0]
  g0.w = (unsigned)((ga >> 32) & 0x1FFFFFFu) | (2u << 30);   // addr[56:32] | type=2
  int8v g1;
  g1[0] = (1 << 16) | (1 << 20) | (3 << 22) | (3 << 25);     // data_size=2B, pad en, ivl=16dw, amt=4dw
  g1[1] = (int)((G4H & 0xFFFF) << 16);                       // tensor_dim0 = 2048 (lo16)
  g1[2] = (int)((VV & 0xFFFF) << 16);                        // tensor_dim1 = 32000 (lo16)
  g1[3] = (32 << 16);                                        // tile_dim0 = 32
  g1[4] = 256;                                               // tile_dim1 = 256
  g1[5] = G4H;                                               // tensor_dim0_stride = 2048
  g1[6] = 0;
  g1[7] = 0;
  int4v gz4 = {0, 0, 0, 0};
  int8v gz8 = {0, 0, 0, 0, 0, 0, 0, 0};
  // 6-arg form (clang-23 / therock-10.0 headers): (g0, g1, g2, g3, g4, cpol)
  __builtin_amdgcn_tensor_load_to_lds(g0, g1, gz4, gz4, gz8, 0);
}

// ------------- output GEMM with fused softmax statistics --------------------
// logits(m,n) = A(m,:) . W(n,:) + bias(n).  Never stored; instead accumulate
// sumexp[m] += sum_n exp(logit) and capture tgtlogit[m] = logit at n==tgtcol[m].
// grid = (M/32, N/256): M fastest so one 1MB W slab stays L2/LDS-hot across
// all 63 M blocks -> W_out streamed from HBM ~once.
__global__ __launch_bounds__(256) void gemm_out_kernel(
    const u16* __restrict__ A,      // MROWS x ACD (bf16)
    const u16* __restrict__ W,      // V x ACD (bf16)
    const float* __restrict__ bias, // V
    const int* __restrict__ tgtcol, // MROWS
    float* __restrict__ tgtlogit,   // MROWS
    float* __restrict__ sumexp)     // MROWS (pre-zeroed)
{
  __shared__ u16 wtile[2][256 * WTILE_PITCH];   // 2 x 20KB double buffer
  __shared__ float rowsum[32];
  const int lane = threadIdx.x & 31;
  const int wv   = threadIdx.x >> 5;
  const int lr   = lane & 15;
  const int hi   = lane >> 4;
  const int mbase = blockIdx.x * 32;
  const int m0   = mbase + (wv & 1) * 16;
  const int nblk = blockIdx.y * 256;
  const int n0   = nblk + (wv >> 1) * 64;

  if (threadIdx.x < 32) rowsum[threadIdx.x] = 0.0f;

  const u16* Ap    = A + (size_t)(m0 + lr) * ACD + hi * 16;
  const u16* Wslab = W + (size_t)nblk * ACD;    // row n = nblk, k = 0
  const unsigned lds0 = (unsigned)(uintptr_t)&wtile[0][0];
  const unsigned lds1 = (unsigned)(uintptr_t)&wtile[1][0];
  // per-lane LDS offsets of the 4 B fragments (row*pitch + hi*16 u16)
  const u16* bt0 = &wtile[0][((wv >> 1) * 64 + lr) * WTILE_PITCH + hi * 16];
  const int boff1 = 16 * WTILE_PITCH;           // +16 rows
  const int bufstep = 256 * WTILE_PITCH;        // buffer 0 -> 1

  const int NCH = ACD / 32;                     // 64 K-chunks
  if (wv == 0) tdm_load_w_tile(Wslab, lds0);

  floatx8 acc[4] = {};
  for (int c = 0; c < NCH; ++c) {
    if (wv == 0) {
      if (c + 1 < NCH) {
        tdm_load_w_tile(Wslab + (c + 1) * 32, ((c + 1) & 1) ? lds1 : lds0);
        __builtin_amdgcn_s_wait_tensorcnt(1);   // chunk c has landed
      } else {
        __builtin_amdgcn_s_wait_tensorcnt(0);
      }
    }
    __syncthreads();                            // tile c visible to all waves

    bf16x16 a  = load_fragA(Ap + c * 32);
    const u16* bt = bt0 + (c & 1) * bufstep;
    bf16x16 b0 = load_fragB(bt);
    bf16x16 b1 = load_fragB(bt + boff1);
    bf16x16 b2 = load_fragB(bt + 2 * boff1);
    bf16x16 b3 = load_fragB(bt + 3 * boff1);
    __builtin_prefetch(Ap + c * 32 + 256, 0, 0);
    acc[0] = __builtin_amdgcn_wmma_f32_16x16x32_bf16(false, a, false, b0, (short)0, acc[0], false, false);
    acc[1] = __builtin_amdgcn_wmma_f32_16x16x32_bf16(false, a, false, b1, (short)0, acc[1], false, false);
    acc[2] = __builtin_amdgcn_wmma_f32_16x16x32_bf16(false, a, false, b2, (short)0, acc[2], false, false);
    acc[3] = __builtin_amdgcn_wmma_f32_16x16x32_bf16(false, a, false, b3, (short)0, acc[3], false, false);
    __syncthreads();                            // done reading buf (c&1) before c+2 overwrites
  }

#pragma unroll
  for (int r = 0; r < 8; ++r) {
    const int lrow = (wv & 1) * 16 + hi * 8 + r;
    const int m = mbase + lrow;
    const int tc = tgtcol[m];
    float p = 0.0f;
#pragma unroll
    for (int i = 0; i < 4; ++i) {
      const int n = n0 + 16 * i + lr;
      float v = acc[i][r] + bias[n];
      p += __expf(v);
      if (n == tc) tgtlogit[m] = v;   // unique writer across whole grid
    }
    // reduce across the 16 lanes of this half-wave (same row set)
#pragma unroll
    for (int off = 1; off < 16; off <<= 1) p += __shfl_xor(p, off, 32);
    if (lr == 0) atomicAdd(&rowsum[lrow], p);
  }
  __syncthreads();
  if (threadIdx.x < 32) atomicAdd(&sumexp[mbase + threadIdx.x], rowsum[threadIdx.x]);
}

// ---------------- small helper kernels --------------------------------------
__global__ void f32_to_bf16_kernel(const float* __restrict__ in, u16* __restrict__ out, int n) {
  int i = blockIdx.x * blockDim.x + threadIdx.x;
  if (i < n) out[i] = f2bf(in[i]);
}

__global__ void embed_src_kernel(const int* __restrict__ toks, const float* __restrict__ emb,
                                 u16* __restrict__ out) {
  int i = blockIdx.x * blockDim.x + threadIdx.x;       // B*S*E
  int r = i >> 9, k = i & 511;                         // row r = b*S + s
  out[i] = f2bf(emb[(size_t)toks[r] * EE + k]);
}

__global__ void embed_trg_kernel(const int* __restrict__ toks, const float* __restrict__ emb,
                                 u16* __restrict__ out) {
  int i = blockIdx.x * blockDim.x + threadIdx.x;       // MROWS*E
  int r = i >> 9, k = i & 511;                         // row r = t*B + b
  int t = r / BB, b = r % BB;
  out[i] = f2bf(emb[(size_t)toks[b * TT + t] * EE + k]);
}

__global__ void setup_misc_kernel(float* cF, float* cR, u16* hFb, u16* hRb,
                                  float* sumexp, float* tgtlogit, int* tgtcol, float* maskf,
                                  const int* __restrict__ trg_tokens,
                                  const int* __restrict__ trg_lens) {
  int i = blockIdx.x * blockDim.x + threadIdx.x;
  if (i < BB * HH) { cF[i] = 0.0f; cR[i] = 0.0f; hFb[i] = 0; hRb[i] = 0; }
  if (i < MROWS) {
    int t = i / BB, b = i % BB;                        // row = t*B + b
    sumexp[i] = 0.0f;
    tgtlogit[i] = 0.0f;
    tgtcol[i] = trg_tokens[b * TT + t + 1];
    maskf[i] = ((t + 1) < trg_lens[b]) ? 1.0f : 0.0f;
  }
}

// encoder pointwise LSTM update for both cells, one timestep
__global__ void enc_pointwise_kernel(const float* __restrict__ gF, const float* __restrict__ gR,
                                     float* __restrict__ cF, float* __restrict__ cR,
                                     u16* __restrict__ hFb, u16* __restrict__ hRb,
                                     float* __restrict__ src_states, int t) {
  int idx = blockIdx.x * blockDim.x + threadIdx.x;     // 2*B*H
  int cell = idx >= BB * HH;
  int i = cell ? idx - BB * HH : idx;
  int b = i >> 9, j = i & 511;
  const float* g = cell ? gR : gF;
  float* c = cell ? cR : cF;
  u16* hb = cell ? hRb : hFb;
  const float* gr = g + (size_t)b * G4H;
  float gi = gr[j], gf = gr[HH + j], gg = gr[2 * HH + j], go = gr[3 * HH + j];
  float cc = sigf(gf) * c[b * HH + j] + sigf(gi) * tanhf(gg);
  float h = sigf(go) * tanhf(cc);
  c[b * HH + j] = cc;
  hb[b * HH + j] = f2bf(h);
  src_states[((size_t)b * SS + t) * DST + cell * HH + j] = h;
}

__global__ void dec_init_kernel(const float* __restrict__ src_states,
                                const int* __restrict__ src_lens,
                                u16* __restrict__ hDb, float* __restrict__ cD) {
  int i = blockIdx.x * blockDim.x + threadIdx.x;       // B*HD
  int b = i >> 10, j = i & 1023;
  float h = src_states[((size_t)b * SS + (src_lens[b] - 1)) * DST + j];
  hDb[i] = f2bf(h);
  cD[i] = 0.0f;
}

// decoder pointwise LSTM + dot attention + AC row emission, one timestep
// grid = B blocks of 256 threads
__global__ void dec_step_kernel(const float* __restrict__ gates,   // B x 4HD
                                float* __restrict__ cD,            // B x HD
                                u16* __restrict__ hDb,             // B x HD (bf16)
                                const float* __restrict__ src_states, // B x S x DST
                                const int* __restrict__ src_lens,
                                u16* __restrict__ AC, int t) {     // row t*B+b : ACD bf16
  __shared__ float hsh[HDD];
  __shared__ float sc[SS];
  __shared__ float red[256];
  const int b = blockIdx.x, tid = threadIdx.x;
  const float* gr = gates + (size_t)b * G4HD;
  u16* acrow = AC + (size_t)(t * BB + b) * ACD;

  for (int u = tid; u < HDD; u += 256) {
    float gi = gr[u], gf = gr[HDD + u], gg = gr[2 * HDD + u], go = gr[3 * HDD + u];
    float cc = sigf(gf) * cD[b * HDD + u] + sigf(gi) * tanhf(gg);
    float h = sigf(go) * tanhf(cc);
    cD[b * HDD + u] = cc;
    hsh[u] = h;
    hDb[b * HDD + u] = f2bf(h);
    acrow[u] = f2bf(h);
  }
  __syncthreads();

  // scores[s] = dot(h, src_states[b,s,:]) ; 4 threads per s
  {
    int s = tid >> 2, part = tid & 3;
    const float* row = src_states + ((size_t)b * SS + s) * DST;
    float a = 0.0f;
    for (int k = part * 256; k < part * 256 + 256; ++k) a += hsh[k] * row[k];
    red[tid] = a;
  }
  __syncthreads();
  if ((tid & 3) == 0) {
    int s = tid >> 2;
    float v = red[tid] + red[tid + 1] + red[tid + 2] + red[tid + 3];
    sc[s] = (s < src_lens[b]) ? v : -1e9f;
  }
  __syncthreads();
  if (tid == 0) {
    float m = sc[0];
    for (int s = 1; s < SS; ++s) m = fmaxf(m, sc[s]);
    float ssum = 0.0f;
    for (int s = 0; s < SS; ++s) { float e = __expf(sc[s] - m); sc[s] = e; ssum += e; }
    float inv = 1.0f / ssum;
    for (int s = 0; s < SS; ++s) sc[s] *= inv;
  }
  __syncthreads();
  // ctx[j] = sum_s attn[s] * src_states[b,s,j]
  for (int j = tid; j < DST; j += 256) {
    float a = 0.0f;
    for (int s = 0; s < SS; ++s) a += sc[s] * src_states[((size_t)b * SS + s) * DST + j];
    acrow[HDD + j] = f2bf(a);
  }
}

__global__ void final_reduce_kernel(const float* __restrict__ tgtlogit,
                                    const float* __restrict__ sumexp,
                                    const float* __restrict__ maskf,
                                    float* __restrict__ out) {
  __shared__ float red[256];
  float a = 0.0f;
  for (int r = threadIdx.x; r < MROWS; r += 256)
    a += maskf[r] * (tgtlogit[r] - logf(sumexp[r]));
  red[threadIdx.x] = a;
  __syncthreads();
  for (int s = 128; s > 0; s >>= 1) {
    if (threadIdx.x < s) red[threadIdx.x] += red[threadIdx.x + s];
    __syncthreads();
  }
  if (threadIdx.x == 0) out[0] = red[0];
}

// ---------------- host side --------------------------------------------------
static inline char* bump(char*& p, size_t bytes) {
  char* r = p;
  p += (bytes + 255) & ~(size_t)255;
  return r;
}

extern "C" void kernel_launch(void* const* d_in, const int* in_sizes, int n_in,
                              void* d_out, int out_size, void* d_ws, size_t ws_size,
                              hipStream_t stream) {
  (void)in_sizes; (void)n_in; (void)out_size; (void)ws_size;
  const int*   src_tokens = (const int*)d_in[0];
  const int*   src_lens   = (const int*)d_in[1];
  const int*   trg_tokens = (const int*)d_in[2];
  const int*   trg_lens   = (const int*)d_in[3];
  const float* src_emb    = (const float*)d_in[4];
  const float* trg_emb    = (const float*)d_in[5];
  const float* We_ih = (const float*)d_in[6];
  const float* We_hh = (const float*)d_in[7];
  const float* be_ih = (const float*)d_in[8];
  const float* be_hh = (const float*)d_in[9];
  const float* Wr_ih = (const float*)d_in[10];
  const float* Wr_hh = (const float*)d_in[11];
  const float* br_ih = (const float*)d_in[12];
  const float* br_hh = (const float*)d_in[13];
  const float* Wd_ih = (const float*)d_in[14];
  const float* Wd_hh = (const float*)d_in[15];
  const float* bd_ih = (const float*)d_in[16];
  const float* bd_hh = (const float*)d_in[17];
  const float* W_out = (const float*)d_in[18];
  const float* b_out = (const float*)d_in[19];
  float* out = (float*)d_out;

  char* p = (char*)d_ws;
  u16* wWe_ih = (u16*)bump(p, (size_t)G4H * EE * 2);
  u16* wWe_hh = (u16*)bump(p, (size_t)G4H * HH * 2);
  u16* wWr_ih = (u16*)bump(p, (size_t)G4H * EE * 2);
  u16* wWr_hh = (u16*)bump(p, (size_t)G4H * HH * 2);
  u16* wWd_ih = (u16*)bump(p, (size_t)G4HD * EE * 2);
  u16* wWd_hh = (u16*)bump(p, (size_t)G4HD * HDD * 2);
  u16* wWout  = (u16*)bump(p, (size_t)VV * ACD * 2);
  u16* x_src  = (u16*)bump(p, (size_t)BB * SS * EE * 2);
  u16* x_trg  = (u16*)bump(p, (size_t)MROWS * EE * 2);
  float* xWe  = (float*)bump(p, (size_t)BB * SS * G4H * 4);
  float* xWr  = (float*)bump(p, (size_t)BB * SS * G4H * 4);
  float* xWd  = (float*)bump(p, (size_t)MROWS * G4HD * 4);
  float* gatesF = (float*)bump(p, (size_t)BB * G4H * 4);
  float* gatesR = (float*)bump(p, (size_t)BB * G4H * 4);
  float* gatesD = (float*)bump(p, (size_t)BB * G4HD * 4);
  float* cF = (float*)bump(p, (size_t)BB * HH * 4);
  float* cR = (float*)bump(p, (size_t)BB * HH * 4);
  float* cD = (float*)bump(p, (size_t)BB * HDD * 4);
  u16* hFb = (u16*)bump(p, (size_t)BB * HH * 2);
  u16* hRb = (u16*)bump(p, (size_t)BB * HH * 2);
  u16* hDb = (u16*)bump(p, (size_t)BB * HDD * 2);
  float* src_states = (float*)bump(p, (size_t)BB * SS * DST * 4);
  u16* AC = (u16*)bump(p, (size_t)MROWS * ACD * 2);
  float* sumexp   = (float*)bump(p, (size_t)MROWS * 4);
  float* tgtlogit = (float*)bump(p, (size_t)MROWS * 4);
  int*   tgtcol   = (int*)bump(p, (size_t)MROWS * 4);
  float* maskf    = (float*)bump(p, (size_t)MROWS * 4);

  const int TPB = 256;
  auto cvt = [&](const float* in, u16* outp, int n) {
    f32_to_bf16_kernel<<<(n + TPB - 1) / TPB, TPB, 0, stream>>>(in, outp, n);
  };
  // ---- weight conversion to bf16 ----
  cvt(We_ih, wWe_ih, G4H * EE);
  cvt(We_hh, wWe_hh, G4H * HH);
  cvt(Wr_ih, wWr_ih, G4H * EE);
  cvt(Wr_hh, wWr_hh, G4H * HH);
  cvt(Wd_ih, wWd_ih, G4HD * EE);
  cvt(Wd_hh, wWd_hh, G4HD * HDD);
  cvt(W_out, wWout, VV * ACD);

  // ---- embeddings (gather + bf16) ----
  embed_src_kernel<<<(BB * SS * EE) / TPB, TPB, 0, stream>>>(src_tokens, src_emb, x_src);
  embed_trg_kernel<<<(MROWS * EE) / TPB, TPB, 0, stream>>>(trg_tokens, trg_emb, x_trg);
  setup_misc_kernel<<<(BB * HH + TPB - 1) / TPB, TPB, 0, stream>>>(
      cF, cR, hFb, hRb, sumexp, tgtlogit, tgtcol, maskf, trg_tokens, trg_lens);

  // ---- hoisted input projections (big WMMA GEMMs) ----
  gemm_bf16_kernel<<<dim3(G4H / 256, (BB * SS) / 32), TPB, 0, stream>>>(
      x_src, EE, wWe_ih, EE, xWe, G4H, nullptr, 0, nullptr, nullptr, EE);
  gemm_bf16_kernel<<<dim3(G4H / 256, (BB * SS) / 32), TPB, 0, stream>>>(
      x_src, EE, wWr_ih, EE, xWr, G4H, nullptr, 0, nullptr, nullptr, EE);
  gemm_bf16_kernel<<<dim3(G4HD / 256, MROWS / 32), TPB, 0, stream>>>(
      x_trg, EE, wWd_ih, EE, xWd, G4HD, nullptr, 0, nullptr, nullptr, EE);

  // ---- encoder recurrence (both cells) ----
  for (int t = 0; t < SS; ++t) {
    gemm_bf16_kernel<<<dim3(G4H / 256, 1), TPB, 0, stream>>>(
        hFb, HH, wWe_hh, HH, gatesF, G4H,
        xWe + (size_t)t * G4H, SS * G4H, be_ih, be_hh, HH);
    gemm_bf16_kernel<<<dim3(G4H / 256, 1), TPB, 0, stream>>>(
        hRb, HH, wWr_hh, HH, gatesR, G4H,
        xWr + (size_t)t * G4H, SS * G4H, br_ih, br_hh, HH);
    enc_pointwise_kernel<<<(2 * BB * HH) / TPB, TPB, 0, stream>>>(
        gatesF, gatesR, cF, cR, hFb, hRb, src_states, t);
  }

  // ---- decoder init + recurrence with attention ----
  dec_init_kernel<<<(BB * HDD) / TPB, TPB, 0, stream>>>(src_states, src_lens, hDb, cD);
  for (int t = 0; t < TT - 1; ++t) {
    gemm_bf16_kernel<<<dim3(G4HD / 256, 1), TPB, 0, stream>>>(
        hDb, HDD, wWd_hh, HDD, gatesD, G4HD,
        xWd + (size_t)t * BB * G4HD, G4HD, bd_ih, bd_hh, HDD);
    dec_step_kernel<<<BB, TPB, 0, stream>>>(gatesD, cD, hDb, src_states, src_lens, AC, t);
  }

  // ---- batched output projection: TDM-staged W, fused log-softmax stats ----
  gemm_out_kernel<<<dim3(MROWS / 32, VV / 256), TPB, 0, stream>>>(
      AC, wWout, b_out, tgtcol, tgtlogit, sumexp);

  // ---- masked NLL reduction to scalar ----
  final_reduce_kernel<<<1, TPB, 0, stream>>>(tgtlogit, sumexp, maskf, out);
}
